// StructureLearner_2559800508645
// MI455X (gfx1250) — compile-verified, matrix operand order checked by reading, not run on hardware
//
#include <hip/hip_runtime.h>

#define D      512    // feature dim (fixed by reference)
#define PITCH  520    // padded LDS pitch in bf16 (1040B: 260 dwords = bank-conflict-free)
#define NFRAG  (D / 32)
#define K1     31     // k+1 = 31 neighbors kept per row
#define KEEP_PROB 0.9f

typedef __attribute__((ext_vector_type(16))) __bf16 v16bf;
typedef __attribute__((ext_vector_type(8)))  float  v8f;
typedef __attribute__((ext_vector_type(4)))  unsigned uint4v;

struct U4 { unsigned int x, y, z, w; };
union Frag { uint4v q[2]; v16bf v; };

// ds_load_b128 pair for one 16-bit WMMA fragment; asm-volatile so issue order
// is preserved (software pipeline stays two fragments deep).
__device__ __forceinline__ void ds_load_frag(Frag& f, unsigned addr, int off)
{
    asm volatile("ds_load_b128 %0, %2 offset:%c3\n\t"
                 "ds_load_b128 %1, %2 offset:%c4"
                 : "=v"(f.q[0]), "=v"(f.q[1])
                 : "v"(addr), "i"(off), "i"(off + 32));
}
// Partial dscnt wait, data-tied to the fragment it guards so the consuming
// WMMA cannot be scheduled above it.
__device__ __forceinline__ void ds_wait_frag(Frag& f, int cnt)
{
    asm volatile("s_wait_dscnt %c2"
                 : "+v"(f.q[0]), "+v"(f.q[1])
                 : "i"(cnt));
}

// ---------------------------------------------------------------------------
// Kernel 1: row L2-normalize x (f32) -> xn (bf16).  One block per row.
// ---------------------------------------------------------------------------
__global__ void __launch_bounds__(256)
rownorm_kernel(const float* __restrict__ x, __bf16* __restrict__ xn)
{
    __shared__ float red[256];
    const int row = blockIdx.x;
    const int t   = threadIdx.x;
    const float* xr = x + (size_t)row * D;

    float v0 = xr[t];
    float v1 = xr[t + 256];
    red[t] = v0 * v0 + v1 * v1;
    __syncthreads();
    for (int off = 128; off > 0; off >>= 1) {
        if (t < off) red[t] += red[t + off];
        __syncthreads();
    }
    const float rs = rsqrtf(red[0]);
    __bf16* o = xn + (size_t)row * D;
    o[t]       = (__bf16)(v0 * rs);
    o[t + 256] = (__bf16)(v1 * rs);
}

// ---------------------------------------------------------------------------
// Kernel 2: fused  sim = xn @ xn^T  (bf16 WMMA, f32 acc)  +  per-row top-31.
//
// Block = 64 output rows, 8 waves = 4 row-tiles x 2 col-groups; each step
// covers 64 rows x 32 cols.  A panel async-DMA'd to LDS once (padded pitch,
// bank-conflict-free) and hoisted into 128 VGPRs per wave.  B panels
// double-buffered through LDS with global_load_async_to_lds_b128; fragment
// reads are asm-pipelined two stages deep with partial dscnt waits.
// ---------------------------------------------------------------------------
__global__ void __launch_bounds__(256)
gemm_topk_kernel(const __bf16* __restrict__ xn,
                 float* __restrict__ topv, int* __restrict__ topi, int n)
{
    __shared__ __bf16 lds_a[64 * PITCH];       // ~65 KB  A row panel (padded)
    __shared__ __bf16 lds_b[2][32 * PITCH];    // 2 x ~32.5 KB B double buffer
    __shared__ float  sim[64][33];             // result tile (+1 col pad)
    __shared__ float  tv[64][33];              // sorted top-31 values (desc)
    __shared__ int    ti[64][33];              // matching column indices

    const int tid  = threadIdx.x;
    const int wave = tid >> 5;
    const int lane = tid & 31;
    const int rt   = wave >> 1;                // row-tile   0..3
    const int cg   = wave & 1;                 // col-group  0..1
    const int rowBase = blockIdx.x * 64;

    const unsigned lbaseA  = (unsigned)(unsigned long long)(void*)lds_a;
    const unsigned lbaseB0 = (unsigned)(unsigned long long)(void*)&lds_b[0][0];
    const unsigned lbaseB1 = (unsigned)(unsigned long long)(void*)&lds_b[1][0];
    const unsigned long long gbase = (unsigned long long)(const void*)xn;

    // ---- Stage A panel (64 rows x 1024B) -> padded LDS via async DMA ----
    {
        const unsigned long long ga =
            gbase + (unsigned long long)rowBase * (D * 2);
        #pragma unroll
        for (int i = 0; i < 16; ++i) {
            const int p    = tid + i * 256;            // 16B piece index
            const int node = p >> 6;                   // 64 pieces per node row
            const int rem  = (p & 63) * 16;
            unsigned           loff = lbaseA + (unsigned)(node * (PITCH * 2) + rem);
            unsigned long long g    = ga + (unsigned long long)p * 16;
            asm volatile("global_load_async_to_lds_b128 %0, %1, off"
                         :: "v"(loff), "v"(g) : "memory");
        }
    }
    // ---- Stage first B panel (cols 0..31) ----
    {
        #pragma unroll
        for (int i = 0; i < 8; ++i) {
            const int p    = tid + i * 256;
            const int node = p >> 6;
            const int rem  = (p & 63) * 16;
            unsigned           loff = lbaseB0 + (unsigned)(node * (PITCH * 2) + rem);
            unsigned long long g    = gbase + (unsigned long long)p * 16;
            asm volatile("global_load_async_to_lds_b128 %0, %1, off"
                         :: "v"(loff), "v"(g) : "memory");
        }
    }
    asm volatile("s_wait_asynccnt 0x0" ::: "memory");
    for (int i = tid; i < 64 * 32; i += 256) {
        tv[i >> 5][i & 31] = -3.4e38f;
        ti[i >> 5][i & 31] = 0;
    }
    __syncthreads();

    const int lrow = lane & 15;                // M (for A) / N (for B) in tile
    const int koff = (lane >> 4) * 8;          // K sub-offset per ISA layout

    // ---- Hoist this wave's A fragments (invariant) into registers ----
    v16bf afrag[NFRAG];
    #pragma unroll
    for (int kk = 0; kk < D; kk += 32) {
        Frag a;
        const U4* ap = reinterpret_cast<const U4*>(
            lds_a + (rt * 16 + lrow) * PITCH + kk + koff);
        const U4 u0 = ap[0];                   // K = kk+koff   .. +7
        const U4 u1 = ap[2];                   // K = kk+16+koff.. +7
        a.q[0] = uint4v{u0.x, u0.y, u0.z, u0.w};
        a.q[1] = uint4v{u1.x, u1.y, u1.z, u1.w};
        afrag[kk >> 5] = a.v;
    }

    int cur = 0;
    for (int cb = 0; cb < n; cb += 32) {
        // Kick off async fill of the other buffer with the next panel
        if (cb + 32 < n) {
            const unsigned           lb = cur ? lbaseB0 : lbaseB1;
            const unsigned long long g  =
                gbase + (unsigned long long)(cb + 32) * (D * 2);
            #pragma unroll
            for (int i = 0; i < 8; ++i) {
                const int p    = tid + i * 256;
                const int node = p >> 6;
                const int rem  = (p & 63) * 16;
                unsigned           loff = lb + (unsigned)(node * (PITCH * 2) + rem);
                unsigned long long ga   = g + (unsigned long long)p * 16;
                asm volatile("global_load_async_to_lds_b128 %0, %1, off"
                             :: "v"(loff), "v"(ga) : "memory");
            }
        }

        // Compute 16x16 tile.  B fragment ds_loads pipelined 2 stages deep:
        // issue frag k+2, wait only for frag k (dscnt<=4), then WMMA k.
        const unsigned baddr = (cur ? lbaseB1 : lbaseB0)
            + (unsigned)((cg * 16 + lrow) * (PITCH * 2) + koff * 2);
        v8f c = {};
        Frag b[3];
        ds_load_frag(b[0], baddr, 0);
        ds_load_frag(b[1], baddr, 64);
        #pragma unroll
        for (int kk = 0; kk < D; kk += 32) {
            const int i = kk >> 5;
            if (i + 2 < NFRAG) ds_load_frag(b[(i + 2) % 3], baddr, (i + 2) * 64);
            ds_wait_frag(b[i % 3],
                         (i + 2 < NFRAG) ? 4 : ((i + 1 < NFRAG) ? 2 : 0));
            c = __builtin_amdgcn_wmma_f32_16x16x32_bf16(
                    false, afrag[i], false, b[i % 3].v, (short)0, c, false, false);
        }

        // C/D layout: lane<16 -> M=r, lane>=16 -> M=8+r ; N = lane&15
        const int col  = cg * 16 + lrow;
        const int row0 = rt * 16 + (lane >> 4) * 8;
        #pragma unroll
        for (int r = 0; r < 8; ++r) sim[row0 + r][col] = c[r];
        __syncthreads();

        // 64 threads maintain sorted top-31 lists, one per row
        if (tid < 64) {
            float mn = tv[tid][K1 - 1];
            for (int c2 = 0; c2 < 32; ++c2) {
                float v = sim[tid][c2];
                if (v > mn) {
                    int pos = K1 - 1;
                    while (pos > 0 && tv[tid][pos - 1] < v) {
                        tv[tid][pos] = tv[tid][pos - 1];
                        ti[tid][pos] = ti[tid][pos - 1];
                        --pos;
                    }
                    tv[tid][pos] = v;
                    ti[tid][pos] = cb + c2;
                    mn = tv[tid][K1 - 1];
                }
            }
        }

        // Next panel must have landed; sim fully consumed before next write
        asm volatile("s_wait_asynccnt 0x0" ::: "memory");
        __syncthreads();
        cur ^= 1;
    }

    for (int i = tid; i < 64 * K1; i += 256) {
        const int r = i / K1, s = i % K1;
        topv[(size_t)(rowBase + r) * K1 + s] = tv[r][s];
        topi[(size_t)(rowBase + r) * K1 + s] = ti[r][s];
    }
}

// ---------------------------------------------------------------------------
// Kernel 3: zero a float buffer (float4 grid-stride)
// ---------------------------------------------------------------------------
__global__ void __launch_bounds__(256)
zero_kernel(float4* __restrict__ p, size_t n4)
{
    size_t i = (size_t)blockIdx.x * blockDim.x + threadIdx.x;
    const size_t stride = (size_t)gridDim.x * blockDim.x;
    const float4 z = make_float4(0.f, 0.f, 0.f, 0.f);
    for (; i < n4; i += stride) p[i] = z;
}

// ---------------------------------------------------------------------------
// Kernel 4: norm[row] += v ; norm[col] += v   (norm = norm_row + norm_col)
// ---------------------------------------------------------------------------
__global__ void __launch_bounds__(256)
accum_norm_kernel(const float* __restrict__ topv, const int* __restrict__ topi,
                  float* __restrict__ norm, int E)
{
    const int e = blockIdx.x * blockDim.x + threadIdx.x;
    if (e >= E) return;
    const int   row = e / K1;
    const int   col = topi[e];
    const float v   = topv[e];
    atomicAdd(&norm[row], v);
    atomicAdd(&norm[col], v);
}

// ---------------------------------------------------------------------------
// Kernel 5: normalize, nan->0, relu, edge dropout, symmetric scatter-add
// ---------------------------------------------------------------------------
__device__ __forceinline__ unsigned hashu(unsigned x)
{
    x ^= x >> 17; x *= 0xed5ad4bbu;
    x ^= x >> 11; x *= 0xac4c1b51u;
    x ^= x >> 15; x *= 0x31848babu;
    x ^= x >> 14; return x;
}

__global__ void __launch_bounds__(256)
scatter_kernel(const float* __restrict__ topv, const int* __restrict__ topi,
               const float* __restrict__ norm, float* __restrict__ adj,
               int n, int E)
{
    const int e = blockIdx.x * blockDim.x + threadIdx.x;
    if (e >= E) return;
    const int row = e / K1;
    const int col = topi[e];
    float w = topv[e] * rsqrtf(norm[row]) * rsqrtf(norm[col]);
    if (!(w == w)) w = 0.f;          // NaN -> 0
    w = fmaxf(w, 0.f);               // ReLU
    const float    scale  = w / KEEP_PROB;
    const unsigned thresh = (unsigned)(KEEP_PROB * 16777216.0f);
    const float w1 = ((hashu(0x9e3779b9u ^ (unsigned)e)       >> 8) < thresh) ? scale : 0.f;
    const float w2 = ((hashu(0x9e3779b9u ^ (unsigned)(e + E)) >> 8) < thresh) ? scale : 0.f;
    atomicAdd(&adj[(size_t)row * n + col], w1);
    atomicAdd(&adj[(size_t)col * n + row], w2);
}

// ---------------------------------------------------------------------------
extern "C" void kernel_launch(void* const* d_in, const int* in_sizes, int n_in,
                              void* d_out, int out_size, void* d_ws, size_t ws_size,
                              hipStream_t stream)
{
    (void)n_in; (void)out_size; (void)ws_size;
    const float* x = (const float*)d_in[0];
    const int n = in_sizes[0] / D;           // 8192
    const int E = n * K1;

    // Workspace layout (256B aligned):  xn bf16 | topv f32 | topi i32 | norm f32
    char* ws = (char*)d_ws;
    size_t o = 0;
    __bf16* xn = (__bf16*)(ws + o);  o += (size_t)n * D * sizeof(__bf16); o = (o + 255) & ~(size_t)255;
    float* topv = (float*)(ws + o);  o += (size_t)E * sizeof(float);      o = (o + 255) & ~(size_t)255;
    int*   topi = (int*)(ws + o);    o += (size_t)E * sizeof(int);        o = (o + 255) & ~(size_t)255;
    float* norm = (float*)(ws + o);
    float* adj  = (float*)d_out;

    rownorm_kernel<<<n, 256, 0, stream>>>(x, xn);
    zero_kernel<<<32, 256, 0, stream>>>((float4*)norm, (size_t)n / 4);
    gemm_topk_kernel<<<n / 64, 256, 0, stream>>>(xn, topv, topi, n);
    accum_norm_kernel<<<(E + 255) / 256, 256, 0, stream>>>(topv, topi, norm, E);
    zero_kernel<<<4096, 256, 0, stream>>>((float4*)adj, (size_t)n * n / 4);
    scatter_kernel<<<(E + 255) / 256, 256, 0, stream>>>(topv, topi, norm, adj, n, E);
}
